// _NeuronGPT2Model_25409026523692
// MI455X (gfx1250) — matrix operand order repaired and checked by reading, not compiled, not run on hardware
//
#include <hip/hip_runtime.h>
#include <hip/hip_bf16.h>

typedef __bf16 bf16;
typedef __attribute__((ext_vector_type(16))) __bf16 v16bf;
typedef __attribute__((ext_vector_type(4)))  __bf16 v4bf;
typedef __attribute__((ext_vector_type(8)))  float  v8f;
typedef __attribute__((ext_vector_type(4)))  float  f4x;

// types matching the async-to-LDS builtin's parameters (from compile probe):
// param0 is a generic `int __attribute__((vector_size(16)))*`
typedef int v4i_g __attribute__((vector_size(16)));
typedef __attribute__((address_space(3))) v4i_g ls_v4i;

#define BB 2
#define SS 1024
#define DD 1024
#define HH 16
#define HDD 64
#define LL 8
#define FFF 4096
#define TT 512
#define MTOT (BB*SS)

union Frag { f4x f[2]; v16bf v; };

#if defined(__has_builtin)
#if __has_builtin(__builtin_amdgcn_global_load_async_to_lds_b128)
#define HAVE_ASYNC_LDS 1
#endif
#endif

__device__ __forceinline__ void wait_async0() {
#if defined(__has_builtin) && __has_builtin(__builtin_amdgcn_s_wait_asynccnt)
    __builtin_amdgcn_s_wait_asynccnt(0);
#else
    asm volatile("s_wait_asynccnt 0" ::: "memory");
#endif
}

__device__ __forceinline__ float siluf(float x) { return x / (1.f + __expf(-x)); }
__device__ __forceinline__ float geluf(float x) {
    const float c = 0.7978845608028654f;
    return 0.5f * x * (1.f + tanhf(c * (x + 0.044715f * x * x * x)));
}

// ---------------- elementwise: hs = inputs_embeds + wpe ----------------
__global__ void k_embed(const float* __restrict__ ie, const float* __restrict__ wpe,
                        float* __restrict__ hs) {
    size_t i = (size_t)blockIdx.x * blockDim.x + threadIdx.x;
    if (i >= (size_t)BB * SS * DD) return;
    hs[i] = ie[i] + wpe[i % ((size_t)SS * DD)];
}

// ---------------- time-step sinusoidal projection ----------------
__global__ void k_tproj(const float* __restrict__ ts, float* __restrict__ tp) {
    int b = blockIdx.x, j = threadIdx.x;           // 512 threads
    float t = ts[b];
    int jj = (j < 256) ? j : j - 256;
    float f = __expf(-logf(10000.f) * (float)jj / 256.f);
    float a = t * f * 1000.f;
    tp[b * TT + j] = (j < 256) ? __cosf(a) : __sinf(a);
}

// ---------------- tiny dense (B=2 rows): out = act(in) @ W + b ----------------
__global__ void k_dense(const float* __restrict__ in, const float* __restrict__ w,
                        const float* __restrict__ bias, float* __restrict__ out,
                        int K, int N, int silu_in, int silu_out) {
    int i = blockIdx.x * blockDim.x + threadIdx.x;
    if (i >= BB * N) return;
    int b = i / N, n = i % N;
    const float* ip = in + (size_t)b * K;
    float s = bias[n];
    for (int k = 0; k < K; ++k) {
        float v = ip[k];
        if (silu_in) v = siluf(v);
        s += v * w[(size_t)k * N + n];
    }
    if (silu_out) s = siluf(s);
    out[i] = s;
}

// ---------------- modbuf[b][6][D] = sst[l] + adaln[b] ----------------
__global__ void k_addmod(const float* __restrict__ sst_l, const float* __restrict__ adaln,
                         float* __restrict__ mod) {
    int i = blockIdx.x * blockDim.x + threadIdx.x;
    if (i >= BB * 6 * DD) return;
    mod[i] = sst_l[i % (6 * DD)] + adaln[i];
}

// ---------------- fmod[b][2][D] = fsst + emb[b] ----------------
__global__ void k_fmod(const float* __restrict__ fsst, const float* __restrict__ emb,
                       float* __restrict__ fmod) {
    int i = blockIdx.x * blockDim.x + threadIdx.x;
    if (i >= BB * 2 * DD) return;
    int d = i % DD, j = (i / DD) % 2, b = i / (2 * DD);
    fmod[i] = fsst[j * DD + d] + emb[b * DD + d];
}

// ---------------- modulated LayerNorm -> bf16 (per-row block) ----------------
__launch_bounds__(256)
__global__ void k_modln(const float* __restrict__ hs, const float* __restrict__ g,
                        const float* __restrict__ bta, const float* __restrict__ shbase,
                        const float* __restrict__ scbase, int bstride,
                        bf16* __restrict__ out) {
    int row = blockIdx.x;                 // 0..2047
    int bb = row >> 10;
    const float* x = hs + (size_t)row * DD;
    __shared__ float r1[8], r2[8], stats[2];
    float s1 = 0.f, s2 = 0.f;
    for (int d = threadIdx.x; d < DD; d += 256) { float v = x[d]; s1 += v; s2 += v * v; }
    for (int o = 16; o; o >>= 1) { s1 += __shfl_down(s1, o, 32); s2 += __shfl_down(s2, o, 32); }
    int wv = threadIdx.x >> 5;
    if ((threadIdx.x & 31) == 0) { r1[wv] = s1; r2[wv] = s2; }
    __syncthreads();
    if (threadIdx.x == 0) {
        float a = 0.f, b2 = 0.f;
        for (int i = 0; i < 8; ++i) { a += r1[i]; b2 += r2[i]; }
        float mean = a / DD;
        float var = b2 / DD - mean * mean;
        stats[0] = mean; stats[1] = rsqrtf(var + 1e-5f);
    }
    __syncthreads();
    float mean = stats[0], rstd = stats[1];
    const float* sh = shbase + (size_t)bb * bstride;
    const float* sc = scbase + (size_t)bb * bstride;
    bf16* op = out + (size_t)row * DD;
    for (int d = threadIdx.x; d < DD; d += 256) {
        float v = (x[d] - mean) * rstd * g[d] + bta[d];
        op[d] = (bf16)(v * (1.f + sc[d]) + sh[d]);
    }
}

// ---------------- RoPE + pack q,k,v into bf16 per-head buffers ----------------
__global__ void k_ropepack(const float* __restrict__ qkv, bf16* __restrict__ Qb,
                           bf16* __restrict__ Kb, bf16* __restrict__ Vb) {
    int i = blockIdx.x * blockDim.x + threadIdx.x;   // B*S*H*32
    if (i >= BB * SS * HH * 32) return;
    int p = i & 31;
    int h = (i >> 5) & (HH - 1);
    int s = (i >> 9) & (SS - 1);
    int b = i >> 19;
    float f = __expf(-(float)p * (logf(10000.f) / 32.f));
    float sn, cs;
    __sincosf((float)s * f, &sn, &cs);
    size_t base = ((size_t)(b * SS + s)) * 3 * DD + h * HDD + 2 * p;
    float qe = qkv[base],          qo = qkv[base + 1];
    float ke = qkv[base + DD],     ko = qkv[base + DD + 1];
    float ve = qkv[base + 2 * DD], vo = qkv[base + 2 * DD + 1];
    size_t ob = ((size_t)((b * HH + h) * SS + s)) * HDD + 2 * p;
    Qb[ob]     = (bf16)(qe * cs - qo * sn);
    Qb[ob + 1] = (bf16)(qe * sn + qo * cs);
    Kb[ob]     = (bf16)(ke * cs - ko * sn);
    Kb[ob + 1] = (bf16)(ke * sn + ko * cs);
    Vb[ob]     = (bf16)ve;
    Vb[ob + 1] = (bf16)vo;
}

// ---------------- tiled bf16 WMMA GEMM, K-step 64, fused epilogues ----------------
// C[M,N] = A[M,K](bf16) * W[K,N](fp32 -> bf16) + bias
// mode 0: fp32 out     mode 1: out = C*gate + res (fp32)     mode 2: out = gelu(C) (bf16)
__launch_bounds__(128)
__global__ void k_gemm(const bf16* __restrict__ A, const float* __restrict__ W,
                       const float* __restrict__ bias, int K, int N, int mode,
                       const float* __restrict__ gate, int gstride,
                       const float* __restrict__ res,
                       float* __restrict__ outf, bf16* __restrict__ outb) {
    __shared__ bf16 As[64 * 64];   // [m][k]
    __shared__ bf16 Bs[64 * 64];   // [n][k]  (transposed weight tile)
    int bm = blockIdx.y * 64, bn = blockIdx.x * 64;
    int t = threadIdx.x;
    int wave = t >> 5, lane = t & 31;
    int mloc = (wave & 1) * 32, nloc = (wave >> 1) * 32;
    int ml = lane & 15, hi = lane >> 4;
    int kb8 = hi * 8, kb16 = hi * 16;

    v8f acc[2][2];
#pragma unroll
    for (int a = 0; a < 2; ++a)
#pragma unroll
        for (int b = 0; b < 2; ++b)
#pragma unroll
            for (int r = 0; r < 8; ++r) acc[a][b][r] = 0.f;

    for (int k0 = 0; k0 < K; k0 += 64) {
        // ---- stage A tile 64x64 bf16: each thread copies 32 contiguous bf16 ----
        {
            int r = t >> 1, c = (t & 1) * 32;
            const bf16* gp = &A[(size_t)(bm + r) * K + k0 + c];
            bf16* lp = &As[r * 64 + c];
#ifdef HAVE_ASYNC_LDS
            v4i_g* g0 = (v4i_g*)gp;                       // generic int4* (param type per probe)
            ls_v4i* l0 = (ls_v4i*)(unsigned)(size_t)lp;   // LDS offset -> addrspace(3) int4*
            __builtin_amdgcn_global_load_async_to_lds_b128(g0,     l0,     0, 0);
            __builtin_amdgcn_global_load_async_to_lds_b128(g0 + 1, l0 + 1, 0, 0);
            __builtin_amdgcn_global_load_async_to_lds_b128(g0 + 2, l0 + 2, 0, 0);
            __builtin_amdgcn_global_load_async_to_lds_b128(g0 + 3, l0 + 3, 0, 0);
#else
            const f4x* g4 = reinterpret_cast<const f4x*>(gp);
            f4x* l4 = reinterpret_cast<f4x*>(lp);
            l4[0] = g4[0]; l4[1] = g4[1]; l4[2] = g4[2]; l4[3] = g4[3];
#endif
        }
        // ---- stage W tile transposed via register transpose: thread owns 4k x 8n ----
        {
            int kb = (t & 15) * 4, nb = (t >> 4) * 8;
            f4x wa[4], wb[4];
#pragma unroll
            for (int j = 0; j < 4; ++j) {
                const float* wp = &W[(size_t)(k0 + kb + j) * N + bn + nb];
                wa[j] = *reinterpret_cast<const f4x*>(wp);
                wb[j] = *reinterpret_cast<const f4x*>(wp + 4);
            }
            if (k0 + 64 < K)
                __builtin_prefetch(&W[(size_t)(k0 + 64 + kb) * N + bn + nb], 0, 0);
#pragma unroll
            for (int n = 0; n < 8; ++n) {
                v4bf pk;
#pragma unroll
                for (int j = 0; j < 4; ++j) {
                    float v = (n < 4) ? wa[j][n] : wb[j][n - 4];
                    pk[j] = (bf16)v;
                }
                *reinterpret_cast<v4bf*>(&Bs[(nb + n) * 64 + kb]) = pk;
            }
        }
#ifdef HAVE_ASYNC_LDS
        wait_async0();
#endif
        __syncthreads();

#pragma unroll
        for (int kc = 0; kc < 64; kc += 32) {
            Frag af[2], bf[2];
#pragma unroll
            for (int i = 0; i < 2; ++i) {
                const bf16* p = &As[(mloc + i * 16 + ml) * 64 + kc];
                af[i].f[0] = *reinterpret_cast<const f4x*>(p + kb8);
                af[i].f[1] = *reinterpret_cast<const f4x*>(p + kb8 + 16);
            }
#pragma unroll
            for (int i = 0; i < 2; ++i) {
                const bf16* p = &Bs[(nloc + i * 16 + ml) * 64 + kc];
                bf[i].f[0] = *reinterpret_cast<const f4x*>(p + kb8);
                bf[i].f[1] = *reinterpret_cast<const f4x*>(p + kb8 + 16);
            }
#pragma unroll
            for (int i = 0; i < 2; ++i)
#pragma unroll
                for (int j = 0; j < 2; ++j)
                    acc[i][j] = __builtin_amdgcn_wmma_f32_16x16x32_bf16(
                        false, af[i].v, false, bf[j].v, (short)0, acc[i][j], false, false);
        }
        __syncthreads();
    }

    // ---- epilogue ----
#pragma unroll
    for (int i = 0; i < 2; ++i) {
#pragma unroll
        for (int j = 0; j < 2; ++j) {
            int ng = bn + nloc + j * 16 + ml;
            int mg0 = bm + mloc + i * 16 + hi * 8;
            float bv = bias[ng];
#pragma unroll
            for (int r = 0; r < 8; ++r) {
                int m = mg0 + r;
                size_t oi = (size_t)m * N + ng;
                float v = acc[i][j][r] + bv;
                if (mode == 0) {
                    outf[oi] = v;
                } else if (mode == 1) {
                    int b = m >> 10;
                    outf[oi] = v * gate[(size_t)b * gstride + ng] + res[oi];
                } else {
                    outb[oi] = (bf16)geluf(v);
                }
            }
        }
    }
}

// ---------------- single-wave flash attention (16 q-rows / block) ----------------
__launch_bounds__(32)
__global__ void k_flash(const bf16* __restrict__ Qb, const bf16* __restrict__ Kb,
                        const bf16* __restrict__ Vb, const float* __restrict__ am,
                        bf16* __restrict__ Ob) {
    __shared__ float scb[16 * 32];
    __shared__ bf16 P[16 * 32];
    __shared__ bf16 Vt[64 * 32];   // [d][key]
    __shared__ float alphas[16], lsums[16];

    int blk = blockIdx.x;
    int qt = blk & (SS / 16 - 1);
    int bh = blk >> 6;
    int b = bh >> 4, h = bh & (HH - 1);
    int q0 = qt * 16;
    int lane = threadIdx.x, ml = lane & 15, hi = lane >> 4;

    const bf16* Qp = Qb + ((size_t)bh * SS) * HDD;
    const bf16* Kp = Kb + ((size_t)bh * SS) * HDD;
    const bf16* Vp = Vb + ((size_t)bh * SS) * HDD;

    Frag qf[2];
#pragma unroll
    for (int c = 0; c < 2; ++c) {
        const bf16* p = &Qp[(size_t)(q0 + ml) * HDD + c * 32 + hi * 8];
        qf[c].f[0] = *reinterpret_cast<const f4x*>(p);
        qf[c].f[1] = *reinterpret_cast<const f4x*>(p + 16);
    }

    float mrow = -1e30f, lrow = 0.f;   // valid on lanes 0..15
    v8f o[4];
#pragma unroll
    for (int nt = 0; nt < 4; ++nt)
#pragma unroll
        for (int r = 0; r < 8; ++r) o[nt][r] = 0.f;

    for (int k0 = 0; k0 < SS; k0 += 32) {
        // stage V tile transposed: Vt[d][key]
        {
            const bf16* vr = &Vp[(size_t)(k0 + lane) * HDD];
#pragma unroll
            for (int d0 = 0; d0 < 64; d0 += 8) {
                union { f4x u; bf16 e[8]; } U;
                U.u = *reinterpret_cast<const f4x*>(vr + d0);
#pragma unroll
                for (int j = 0; j < 8; ++j) Vt[(d0 + j) * 32 + lane] = U.e[j];
            }
        }
        // scores: 16 q x 32 keys
        v8f s0, s1;
#pragma unroll
        for (int r = 0; r < 8; ++r) { s0[r] = 0.f; s1[r] = 0.f; }
#pragma unroll
        for (int c = 0; c < 2; ++c) {
            Frag kf0, kf1;
            {
                const bf16* p = &Kp[(size_t)(k0 + ml) * HDD + c * 32 + hi * 8];
                kf0.f[0] = *reinterpret_cast<const f4x*>(p);
                kf0.f[1] = *reinterpret_cast<const f4x*>(p + 16);
            }
            {
                const bf16* p = &Kp[(size_t)(k0 + 16 + ml) * HDD + c * 32 + hi * 8];
                kf1.f[0] = *reinterpret_cast<const f4x*>(p);
                kf1.f[1] = *reinterpret_cast<const f4x*>(p + 16);
            }
            s0 = __builtin_amdgcn_wmma_f32_16x16x32_bf16(false, qf[c].v, false, kf0.v,
                                                         (short)0, s0, false, false);
            s1 = __builtin_amdgcn_wmma_f32_16x16x32_bf16(false, qf[c].v, false, kf1.v,
                                                         (short)0, s1, false, false);
        }
        float mk0 = (1.f - am[(size_t)b * SS + k0 + ml]) * -3.4028235e38f;
        float mk1 = (1.f - am[(size_t)b * SS + k0 + 16 + ml]) * -3.4028235e38f;
#pragma unroll
        for (int r = 0; r < 8; ++r) {
            scb[(hi * 8 + r) * 32 + ml]      = s0[r] * 0.125f + mk0;
            scb[(hi * 8 + r) * 32 + 16 + ml] = s1[r] * 0.125f + mk1;
        }
        __syncthreads();

        if (lane < 16) {
            const float* srow = &scb[lane * 32];
            float mmax = mrow;
#pragma unroll
            for (int j = 0; j < 32; ++j) mmax = fmaxf(mmax, srow[j]);
            float alpha = __expf(mrow - mmax);
            float ssum = 0.f;
#pragma unroll
            for (int j = 0; j < 32; ++j) {
                float p = __expf(srow[j] - mmax);
                P[lane * 32 + j] = (bf16)p;
                ssum += p;
            }
            lrow = lrow * alpha + ssum;
            mrow = mmax;
            alphas[lane] = alpha;
        }
        __syncthreads();

        float av[8];
#pragma unroll
        for (int r = 0; r < 8; ++r) av[r] = alphas[hi * 8 + r];
#pragma unroll
        for (int nt = 0; nt < 4; ++nt)
#pragma unroll
            for (int r = 0; r < 8; ++r) o[nt][r] *= av[r];

        Frag pf;
        {
            const bf16* p = &P[ml * 32 + hi * 8];
            pf.f[0] = *reinterpret_cast<const f4x*>(p);
            pf.f[1] = *reinterpret_cast<const f4x*>(p + 16);
        }
#pragma unroll
        for (int nt = 0; nt < 4; ++nt) {
            Frag vf;
            const bf16* p = &Vt[(nt * 16 + ml) * 32 + hi * 8];
            vf.f[0] = *reinterpret_cast<const f4x*>(p);
            vf.f[1] = *reinterpret_cast<const f4x*>(p + 16);
            o[nt] = __builtin_amdgcn_wmma_f32_16x16x32_bf16(false, pf.v, false, vf.v,
                                                            (short)0, o[nt], false, false);
        }
        __syncthreads();
    }

    if (lane < 16) lsums[lane] = lrow;
    __syncthreads();
    float lv[8];
#pragma unroll
    for (int r = 0; r < 8; ++r) lv[r] = 1.f / lsums[hi * 8 + r];
#pragma unroll
    for (int nt = 0; nt < 4; ++nt)
#pragma unroll
        for (int r = 0; r < 8; ++r) {
            size_t oi = ((size_t)(b * SS + q0 + hi * 8 + r)) * DD + h * HDD + nt * 16 + ml;
            Ob[oi] = (bf16)(o[nt][r] * lv[r]);
        }
}

// ============================ host launch ============================
extern "C" void kernel_launch(void* const* d_in, const int* in_sizes, int n_in,
                              void* d_out, int out_size, void* d_ws, size_t ws_size,
                              hipStream_t stream) {
    const float* ie   = (const float*)d_in[0];
    const float* am   = (const float*)d_in[1];
    const float* ts   = (const float*)d_in[2];
    const float* wpe  = (const float*)d_in[3];
    const float* ln1g = (const float*)d_in[4];
    const float* ln1b = (const float*)d_in[5];
    const float* aw   = (const float*)d_in[6];
    const float* ab   = (const float*)d_in[7];
    const float* pw   = (const float*)d_in[8];
    const float* pb   = (const float*)d_in[9];
    const float* ln2g = (const float*)d_in[10];
    const float* ln2b = (const float*)d_in[11];
    const float* fw   = (const float*)d_in[12];
    const float* fb   = (const float*)d_in[13];
    const float* qw2  = (const float*)d_in[14];
    const float* qb2  = (const float*)d_in[15];
    const float* sst  = (const float*)d_in[16];
    const float* tw1  = (const float*)d_in[17];
    const float* tb1  = (const float*)d_in[18];
    const float* tw2  = (const float*)d_in[19];
    const float* tb2  = (const float*)d_in[20];
    const float* adw  = (const float*)d_in[21];
    const float* adb  = (const float*)d_in[22];
    const float* lnfg = (const float*)d_in[23];
    const float* lnfb = (const float*)d_in[24];
    const float* fsst = (const float*)d_in[25];
    const float* ow   = (const float*)d_in[26];
    const float* obv  = (const float*)d_in[27];

    char* w = (char*)d_ws;
    float* hs    = (float*)(w + 0);                         // 8 MiB
    bf16*  xb    = (bf16*) (w + (8ull  << 20));             // 4 MiB
    float* qkv   = (float*)(w + (12ull << 20));             // 24 MiB
    bf16*  Qb    = (bf16*) (w + (36ull << 20));             // 4 MiB
    bf16*  Kb    = (bf16*) (w + (40ull << 20));             // 4 MiB
    bf16*  Vb    = (bf16*) (w + (44ull << 20));             // 4 MiB
    bf16*  Ob    = (bf16*) (w + (48ull << 20));             // 4 MiB
    bf16*  fca   = (bf16*) (w + (52ull << 20));             // 16 MiB
    float* tp    = (float*)(w + (68ull << 20));             // 4 KiB
    float* h1    = (float*)(w + (68ull << 20) + (16 << 10));
    float* emb   = (float*)(w + (68ull << 20) + (32 << 10));
    float* adaln = (float*)(w + (68ull << 20) + (64 << 10));
    float* mod   = (float*)(w + (68ull << 20) + (128 << 10));
    float* fmod  = (float*)(w + (68ull << 20) + (192 << 10));

    // embeddings + time MLP + adaLN
    k_embed<<<(BB * SS * DD + 255) / 256, 256, 0, stream>>>(ie, wpe, hs);
    k_tproj<<<BB, TT, 0, stream>>>(ts, tp);
    k_dense<<<(BB * DD + 255) / 256, 256, 0, stream>>>(tp, tw1, tb1, h1, TT, DD, 0, 1);
    k_dense<<<(BB * DD + 255) / 256, 256, 0, stream>>>(h1, tw2, tb2, emb, DD, DD, 0, 0);
    k_dense<<<(BB * 6 * DD + 255) / 256, 256, 0, stream>>>(emb, adw, adb, adaln, DD, 6 * DD, 1, 0);
    k_fmod<<<(BB * 2 * DD + 255) / 256, 256, 0, stream>>>(fsst, emb, fmod);

    for (int l = 0; l < LL; ++l) {
        const float* sstl = sst + (size_t)l * 6 * DD;
        k_addmod<<<(BB * 6 * DD + 255) / 256, 256, 0, stream>>>(sstl, adaln, mod);

        k_modln<<<MTOT, 256, 0, stream>>>(hs, ln1g + (size_t)l * DD, ln1b + (size_t)l * DD,
                                          mod + 0 * DD, mod + 1 * DD, 6 * DD, xb);
        k_gemm<<<dim3(3 * DD / 64, MTOT / 64), 128, 0, stream>>>(
            xb, aw + (size_t)l * DD * 3 * DD, ab + (size_t)l * 3 * DD,
            DD, 3 * DD, 0, nullptr, 0, nullptr, qkv, nullptr);
        k_ropepack<<<(BB * SS * HH * 32 + 255) / 256, 256, 0, stream>>>(qkv, Qb, Kb, Vb);
        k_flash<<<BB * HH * (SS / 16), 32, 0, stream>>>(Qb, Kb, Vb, am, Ob);
        k_gemm<<<dim3(DD / 64, MTOT / 64), 128, 0, stream>>>(
            Ob, pw + (size_t)l * DD * DD, pb + (size_t)l * DD,
            DD, DD, 1, mod + 2 * DD, 6 * DD, hs, hs, nullptr);
        k_modln<<<MTOT, 256, 0, stream>>>(hs, ln2g + (size_t)l * DD, ln2b + (size_t)l * DD,
                                          mod + 3 * DD, mod + 4 * DD, 6 * DD, xb);
        k_gemm<<<dim3(FFF / 64, MTOT / 64), 128, 0, stream>>>(
            xb, fw + (size_t)l * DD * FFF, fb + (size_t)l * FFF,
            DD, FFF, 2, nullptr, 0, nullptr, nullptr, fca);
        k_gemm<<<dim3(DD / 64, MTOT / 64), 128, 0, stream>>>(
            fca, qw2 + (size_t)l * FFF * DD, qb2 + (size_t)l * DD,
            FFF, DD, 1, mod + 5 * DD, 6 * DD, hs, hs, nullptr);
    }

    k_modln<<<MTOT, 256, 0, stream>>>(hs, lnfg, lnfb, fmod + 0 * DD, fmod + 1 * DD, 2 * DD, xb);
    k_gemm<<<dim3(DD / 64, MTOT / 64), 128, 0, stream>>>(
        xb, ow, obv, DD, DD, 0, nullptr, 0, nullptr, (float*)d_out, nullptr);
}
